// EALSTM_10187662426257
// MI455X (gfx1250) — matrix operand order repaired and checked by reading, not compiled, not run on hardware
//
#include <hip/hip_runtime.h>

typedef _Float16 v16h __attribute__((ext_vector_type(16)));
typedef _Float16 v8h  __attribute__((ext_vector_type(8)));
typedef float    v8f  __attribute__((ext_vector_type(8)));

#define B_  512
#define T_  365
#define DD_ 5
#define DS_ 27
#define H_  256
#define KP  264   // padded LDS row stride in halves (16B-aligned, bank-spread)
#define NTHREADS 512   // 16 wave32 waves -> 4 waves/SIMD for latency hiding

// Dynamic LDS layout (byte offsets)
#define WTG_OFF 0
#define WTO_OFF (H_ * KP * 2)                  // 135168
#define HB_OFF  (2 * H_ * KP * 2)              // 270336
#define W5_OFF  (HB_OFF + 16 * KP * 2)         // 278784  (2 x [5][256] f32)
#define XS_OFF  (W5_OFF + 2 * DD_ * H_ * 4)    // 289024  ([16][8] f32)
#define SMEM_BYTES (XS_OFF + 16 * 8 * 4)       // 289536  (< 320KB per WGP)

__device__ __forceinline__ float fast_sigmoid(float x) {
    return 1.0f / (1.0f + __expf(-x));
}
__device__ __forceinline__ float fast_tanh(float x) {
    return 2.0f / (1.0f + __expf(-2.0f * x)) - 1.0f;
}

// A fragment: h tile [16 x 32] f16 per ISA 16-bit A layout.
// lanes 0-15: M=lane, halves 0..7 = K=k0..k0+7, halves 8..15 = K=k0+16..k0+23
// lanes 16-31: M=lane-16, K offset +8.  => two contiguous 16B LDS reads.
__device__ __forceinline__ v16h load_a_frag(const _Float16* __restrict__ hb,
                                            int hl, int hi, int k0) {
    const _Float16* base = hb + hl * KP + k0 + hi * 8;
    v8h lo = *(const v8h*)(base);
    v8h hh = *(const v8h*)(base + 16);
    v16h a;
#pragma unroll
    for (int j = 0; j < 8; ++j) { a[j] = lo[j]; a[j + 8] = hh[j]; }
    return a;
}

// B fragment: W^T rows are n, contiguous k. lanes 0-15: n=n0+lane, halves 0..7 = K=k0..k0+7,
// halves 8..15 = K=k0+16..k0+23; lanes 16-31: n=n0+lane-16, K offset +8. => two 16B reads.
__device__ __forceinline__ v16h load_b_frag(const _Float16* __restrict__ wt,
                                            int n0, int hl, int hi, int k0) {
    const _Float16* base = wt + (n0 + hl) * KP + k0 + hi * 8;
    v8h lo = *(const v8h*)(base);
    v8h hh = *(const v8h*)(base + 16);
    v16h b;
#pragma unroll
    for (int j = 0; j < 8; ++j) { b[j] = lo[j]; b[j + 8] = hh[j]; }
    return b;
}

__global__ __launch_bounds__(NTHREADS, 1)
void ealstm_fused_kernel(const float* __restrict__ xdyn,   // [B,T,DD]
                         const float* __restrict__ xstat,  // [B,DS]
                         const float* __restrict__ Wi, const float* __restrict__ bi,
                         const float* __restrict__ Wf, const float* __restrict__ bf,
                         const float* __restrict__ Wg, const float* __restrict__ bg,
                         const float* __restrict__ Wo, const float* __restrict__ bo,
                         const float* __restrict__ Wh, const float* __restrict__ bh,
                         float* __restrict__ out)          // [B,1]
{
    extern __shared__ char smem[];
    _Float16* WtG = (_Float16*)(smem + WTG_OFF);   // [256][KP]: WtG[n][k] = Wg[5+k][n]
    _Float16* WtO = (_Float16*)(smem + WTO_OFF);   // [256][KP]: WtO[n][k] = Wo[5+k][n]
    _Float16* Hb  = (_Float16*)(smem + HB_OFF);    // [16][KP]:  h tile, f16
    float*    W5  = (float*)(smem + W5_OFF);       // [0..4][n]=Wg[d][n], [5..9][n]=Wo[d][n]
    float*    XS  = (float*)(smem + XS_OFF);       // [16][8]: x_t tile
    float*    RED = W5;                            // reuse for final reduction

    const int tid  = threadIdx.x;
    const int wave = tid >> 5;       // 0..15
    const int lane = tid & 31;
    const int hl   = lane & 15;
    const int hi   = lane >> 4;
    const int b0   = blockIdx.x * 16;

    // ---- one-time: stage recurrent weights to LDS as f16, transposed [n][k] ----
    // threads 0..255 stage WtG, threads 256..511 stage WtO (coalesced row reads).
    {
        const int n = tid & (H_ - 1);
        _Float16*   dst = (tid < H_) ? WtG : WtO;
        const float* src = (tid < H_) ? Wg : Wo;
        for (int k = 0; k < H_; ++k)
            dst[n * KP + k] = (_Float16)src[(DD_ + k) * H_ + n];
#pragma unroll
        for (int d = 0; d < DD_; ++d) {
            if (tid < H_) W5[d * H_ + n]          = src[d * H_ + n];
            else          W5[(DD_ + d) * H_ + n]  = src[d * H_ + n];
        }
        if (tid < H_) {
#pragma unroll
            for (int m = 0; m < 16; ++m) Hb[m * KP + n] = (_Float16)0.0f;
        }
    }

    // ---- one-time: static gates i,f in registers (D-tile layout) ----
    // wave owns one n-tile: n0 = wave*16; lane column n = n0 + hl; VGPR j -> m = j + hi*8
    const int n0 = wave * 16;
    const int n  = n0 + hl;
    float iG[8], fG[8], cS[8];
    const float bgv = bg[n];
    const float bov = bo[n];
    {
        float ai[8], af[8];
#pragma unroll
        for (int j = 0; j < 8; ++j) { ai[j] = bi[n]; af[j] = bf[n]; }
        for (int d = 0; d < DS_; ++d) {
            const float wi = Wi[d * H_ + n];
            const float wf = Wf[d * H_ + n];
#pragma unroll
            for (int j = 0; j < 8; ++j) {
                const float xv = xstat[(size_t)(b0 + j + hi * 8) * DS_ + d];
                ai[j] += xv * wi;
                af[j] += xv * wf;
            }
        }
#pragma unroll
        for (int j = 0; j < 8; ++j) {
            iG[j] = fast_sigmoid(ai[j]);
            fG[j] = fast_sigmoid(af[j]);
            cS[j] = 0.0f;
        }
    }
    __syncthreads();

    // ---- recurrence over T steps ----
    for (int t = 0; t < T_; ++t) {
        // stage x_t tile (16 rows x 5 features)
        if (tid < 16 * DD_) {
            const int m = tid / DD_, d = tid % DD_;
            XS[m * 8 + d] = xdyn[((size_t)(b0 + m) * T_ + t) * DD_ + d];
        }
        __syncthreads();   // x_t ready AND previous step's h writes visible

        // C-operand init: bias + x_t @ W[:5] (VALU, co-executes with WMMA below)
        float xv[DD_][8];
#pragma unroll
        for (int d = 0; d < DD_; ++d)
#pragma unroll
            for (int j = 0; j < 8; ++j) xv[d][j] = XS[(j + hi * 8) * 8 + d];

        v8f accG, accO;
#pragma unroll
        for (int j = 0; j < 8; ++j) { accG[j] = bgv; accO[j] = bov; }
#pragma unroll
        for (int d = 0; d < DD_; ++d) {
            const float wg5 = W5[d * H_ + n];
            const float wo5 = W5[(DD_ + d) * H_ + n];
#pragma unroll
            for (int j = 0; j < 8; ++j) {
                accG[j] += xv[d][j] * wg5;
                accO[j] += xv[d][j] * wo5;
            }
        }

        // recurrent matmuls: h[16,256] x W[256,256] via WMMA f32_16x16x32_f16.
        // Per wave: one 16x16 output tile for g and one for o; software-pipelined
        // fragment prefetch, with 4 waves/SIMD hiding the residual ds latency.
        {
            v16h a  = load_a_frag(Hb, hl, hi, 0);
            v16h bG = load_b_frag(WtG, n0, hl, hi, 0);
            v16h bO = load_b_frag(WtO, n0, hl, hi, 0);
#pragma unroll
            for (int k0 = 0; k0 < H_; k0 += 32) {
                v16h aN, bGN, bON;
                if (k0 + 32 < H_) {
                    aN  = load_a_frag(Hb, hl, hi, k0 + 32);
                    bGN = load_b_frag(WtG, n0, hl, hi, k0 + 32);
                    bON = load_b_frag(WtO, n0, hl, hi, k0 + 32);
                }
                accG = __builtin_amdgcn_wmma_f32_16x16x32_f16(
                    false, a, false, bG, (short)0, accG, false, false);
                accO = __builtin_amdgcn_wmma_f32_16x16x32_f16(
                    false, a, false, bO, (short)0, accO, false, false);
                a = aN; bG = bGN; bO = bON;
            }
        }
        __syncthreads();   // all waves finished reading Hb before it is rewritten

        // elementwise gate math + state update; write h_next (f16) to LDS
#pragma unroll
        for (int j = 0; j < 8; ++j) {
            const float g = fast_tanh(accG[j]);
            const float o = fast_sigmoid(accO[j]);
            const float c = fG[j] * cS[j] + iG[j] * g;
            cS[j] = c;
            const float h = o * fast_tanh(c);
            const int m = j + hi * 8;
            Hb[m * KP + n] = (_Float16)h;
        }
        // next iteration's leading __syncthreads() publishes these writes
    }
    __syncthreads();

    // ---- final projection: out[b] = h[b,:] @ Wh + bh ----
    if (tid < 256) {
        const int m = tid >> 4, seg = tid & 15;
        float part = 0.0f;
#pragma unroll
        for (int j = 0; j < 16; ++j) {
            const int nn = seg * 16 + j;
            part += (float)Hb[m * KP + nn] * Wh[nn];
        }
        RED[m * 16 + seg] = part;
    }
    __syncthreads();
    if (tid < 16) {
        float s = bh[0];
#pragma unroll
        for (int j = 0; j < 16; ++j) s += RED[tid * 16 + j];
        out[b0 + tid] = s;
    }
}

extern "C" void kernel_launch(void* const* d_in, const int* in_sizes, int n_in,
                              void* d_out, int out_size, void* d_ws, size_t ws_size,
                              hipStream_t stream) {
    (void)in_sizes; (void)n_in; (void)d_ws; (void)ws_size; (void)out_size;

    const float* xdyn  = (const float*)d_in[0];
    const float* xstat = (const float*)d_in[1];
    const float* Wi    = (const float*)d_in[2];
    const float* bi    = (const float*)d_in[3];
    const float* Wf    = (const float*)d_in[4];
    const float* bf    = (const float*)d_in[5];
    const float* Wg    = (const float*)d_in[6];
    const float* bg    = (const float*)d_in[7];
    const float* Wo    = (const float*)d_in[8];
    const float* bo    = (const float*)d_in[9];
    const float* Wh    = (const float*)d_in[10];
    const float* bh    = (const float*)d_in[11];
    float* out = (float*)d_out;

    // Allow >default dynamic LDS (idempotent, not a stream op).
    (void)hipFuncSetAttribute((const void*)ealstm_fused_kernel,
                              hipFuncAttributeMaxDynamicSharedMemorySize,
                              SMEM_BYTES);

    dim3 grid(B_ / 16);     // 32 workgroups, one per WGP, 16 batch rows each
    dim3 block(NTHREADS);   // 16 wave32 waves -> 4 waves per SIMD32
    ealstm_fused_kernel<<<grid, block, SMEM_BYTES, stream>>>(
        xdyn, xstat, Wi, bi, Wf, bf, Wg, bg, Wo, bo, Wh, bh, out);
}
